// Trans_infer_rnn_tree_51539607552426
// MI455X (gfx1250) — matrix operand order repaired and checked by reading, not compiled, not run on hardware
//
#include <hip/hip_runtime.h>
#include <hip/hip_bf16.h>

// ---------------- types / helpers ----------------
typedef __attribute__((ext_vector_type(16))) __bf16 bf16x16;
typedef __attribute__((ext_vector_type(8)))  float  f32x8;

union Frag { uint4 u[2]; bf16x16 v; };

__device__ __forceinline__ uint4 z4() { uint4 r; r.x=0u; r.y=0u; r.z=0u; r.w=0u; return r; }
__device__ __forceinline__ float sigm(float x){ return 1.0f/(1.0f + __expf(-x)); }

// sizes
#define BB   8
#define HH   96
#define WW   96
#define HW   9216          // 96*96
#define CIN  256
#define SPB  2359296       // 9216*256  (per-batch NHWC elems)

// d_out regions (float elems)
#define O_HALF_H   0
#define O_HALF2_H  4423680
#define O_FULL_H   5898240
#define O_HALF_C   6635520
#define O_HALF2_C  11059200
#define O_FULL_C   12533760

// ---------------- NCHW fp32 -> NHWC bf16 ----------------
__global__ void nchw_to_nhwc_bf16(const float* __restrict__ in, __bf16* __restrict__ out, int total) {
  int idx = blockIdx.x * blockDim.x + threadIdx.x;
  if (idx >= total) return;
  int c  = idx & 255;
  int s  = idx >> 8;            // b*9216 + y*96 + x
  int sp = s % HW;
  int b  = s / HW;
  out[idx] = (__bf16)in[((size_t)b * CIN + c) * HW + sp];
}

// ---------------- weight pack: [tap][c][o] bf16 ----------------
// mode 0: leaf [240,256,3,3]; mode 1: h-path (gate_up|f_up|gate_lo|f_lo)[.,266,3,3] x-parts (80 out)
// mode 2: f-path (gate_full|f_full) x-parts, pad to 48 out
__global__ void pack_plain(const float* __restrict__ w0, const float* __restrict__ w1,
                           const float* __restrict__ w2, const float* __restrict__ w3,
                           __bf16* __restrict__ out, int cout, int mode, int total) {
  int idx = blockIdx.x * blockDim.x + threadIdx.x;
  if (idx >= total) return;
  int o   = idx % cout;
  int c   = (idx / cout) % CIN;
  int tap = idx / (cout * CIN);
  float v = 0.0f;
  if (mode == 0) {
    v = w0[((size_t)o * CIN + c) * 9 + tap];
  } else if (mode == 1) {
    const float* w; int oo;
    if      (o < 30) { w = w0; oo = o;      }
    else if (o < 40) { w = w1; oo = o - 30; }
    else if (o < 70) { w = w2; oo = o - 40; }
    else             { w = w3; oo = o - 70; }
    v = w[((size_t)oo * 266 + c) * 9 + tap];
  } else {
    if      (o < 30) v = w0[((size_t)o * 266 + c) * 9 + tap];
    else if (o < 40) v = w1[((size_t)(o - 30) * 266 + c) * 9 + tap];
    else             v = 0.0f;
  }
  out[(size_t)(tap * CIN + c) * cout + o] = (__bf16)v;
}

// ---------------- repack into WMMA B-fragment order ----------------
// out layout: [tap][kc(8)][nt][lane(32)][j(16)]
// B 32x16 bf16 layout: lane (0-15): col N=lane, K={0..7,16..23} ; lane (16-31): col N=lane-16, K={8..15,24..31}
__global__ void repack_bfrag(const __bf16* __restrict__ wp, __bf16* __restrict__ out,
                             int NT, int cout, int total) {
  int idx = blockIdx.x * blockDim.x + threadIdx.x;
  if (idx >= total) return;
  int j    = idx & 15;
  int lane = (idx >> 4) & 31;
  int nt   = (idx >> 9) % NT;
  int kc   = (idx / (512 * NT)) & 7;
  int tap  = idx / (4096 * NT);
  int k = kc * 32 + (j < 8 ? (lane >> 4) * 8 + j : 16 + (lane >> 4) * 8 + (j - 8));
  int o = nt * 16 + (lane & 15);
  out[idx] = wp[(size_t)(tap * CIN + k) * cout + o];
}

// ---------------- implicit-GEMM 3x3 conv with bf16 WMMA ----------------
// act: NHWC bf16 [8,96,96,256] ; wpk: packed B frags ; out: [b][y][x][cout] fp32
// wave = one 16-position M tile (16 consecutive x) x NB N-tiles; K = 9 taps x 256 ch
template<int NB>
__global__ __launch_bounds__(256) void conv_wmma(const __bf16* __restrict__ act,
                                                 const __bf16* __restrict__ wpk,
                                                 float* __restrict__ out,
                                                 int NT, int cout) {
  const int lane  = threadIdx.x & 31;
  const int wave  = threadIdx.x >> 5;
  const int mtile = blockIdx.x * 8 + wave;     // 4608 tiles total
  const int b  = mtile / 576;
  const int r  = mtile % 576;
  const int y  = r / 6;
  const int x0 = (r % 6) * 16;
  const int nt0 = blockIdx.y * NB;
  const int lm = lane & 15, half = lane >> 4;

  f32x8 acc[NB] = {};

  for (int tap = 0; tap < 9; ++tap) {
    const int ky = tap / 3, kx = tap - ky * 3;
    const int iy = y + ky - 1;
    const int ix = x0 + lm + kx - 1;
    const bool ok = (iy >= 0) & (iy < HH) & (ix >= 0) & (ix < WW);
    const int iyc = ok ? iy : 0;
    const int ixc = ok ? ix : 0;
    // A layout: lane<16 holds K {c0..c0+7, c0+16..c0+23}; lane>=16 offset +8. Both contiguous runs.
    const __bf16* arow = act + ((size_t)(b * HH + iyc) * WW + ixc) * CIN + half * 8;
    const __bf16* wtap = wpk + (size_t)(tap * 8) * NT * 512;
    for (int kc = 0; kc < 8; ++kc) {
      Frag a;
      a.u[0] = *(const uint4*)(arow + kc * 32);
      a.u[1] = *(const uint4*)(arow + kc * 32 + 16);
      if (!ok) { a.u[0] = z4(); a.u[1] = z4(); }
      const __bf16* wp = wtap + ((size_t)kc * NT + nt0) * 512 + lane * 16;
#pragma unroll
      for (int nb = 0; nb < NB; ++nb) {
        Frag bf;
        bf.u[0] = *(const uint4*)(wp + nb * 512);
        bf.u[1] = *(const uint4*)(wp + nb * 512 + 8);
        acc[nb] = __builtin_amdgcn_wmma_f32_16x16x32_bf16(
            false, a.v, false, bf.v, (short)0, acc[nb], false, false);
      }
    }
  }
  // C/D layout: lane -> col n=lane%16 ; vgpr rr -> row m=rr+8*(lane/16)
  float* ob = out + ((size_t)(b * HH + y) * WW + x0) * cout;
#pragma unroll
  for (int nb = 0; nb < NB; ++nb) {
    const int n = (nt0 + nb) * 16 + lm;
#pragma unroll
    for (int rr = 0; rr < 8; ++rr)
      ob[(size_t)(rr + half * 8) * cout + n] = acc[nb][rr];
  }
}

// ---------------- leaf gate epilogue ----------------
__global__ void leaf_epilogue(const float* __restrict__ G, float* __restrict__ out, int total) {
  int idx = blockIdx.x * blockDim.x + threadIdx.x;
  if (idx >= total) return;
  int x = idx % 96;
  int y = (idx / 96) % 96;
  int h = (idx / HW) % 10;
  int n = (idx / 92160) % 6;
  int b = idx / 552960;
  size_t base = ((size_t)(b * HH + y) * WW + x) * 240 + n * 40;
  float gi = G[base + h];
  float go = G[base + 20 + h];
  float gg = G[base + 30 + h];
  float c  = sigm(gi) * tanhf(gg);
  float hh = sigm(go) * tanhf(c);
  out[O_HALF_H + idx] = hh;
  out[O_HALF_C + idx] = c;
}

// ---------------- sum of children h fields ----------------
__global__ void sum_children(const float* __restrict__ in, float* __restrict__ out,
                             int nchild, int total) {
  int idx = blockIdx.x * blockDim.x + threadIdx.x;
  if (idx >= total) return;
  int sp = idx % 92160;
  int b  = idx / 92160;
  const float* p = in + (size_t)b * 552960 + sp;
  float s = 0.0f;
  for (int j = 0; j < nchild; ++j) s += p[(size_t)j * 92160];
  out[idx] = s;
}

// ---------------- tree-cell epilogue (small 10-ch convs on VALU) ----------------
__global__ __launch_bounds__(256) void tree_epilogue(
    const float* __restrict__ G, int gstride, int gate_off, int f_off,
    const float* __restrict__ gate_w, const float* __restrict__ f_w,
    const float* __restrict__ hsum, long long hsum_bs, float hsum_scale,
    const float* h0, long long h0bs, const float* h1, long long h1bs,
    const float* h2, long long h2bs, const float* h3, long long h3bs,
    const float* c0, long long c0bs, const float* c1, long long c1bs,
    const float* c2, long long c2bs, const float* c3, long long c3bs,
    int nchild,
    float* hout0, long long ho0bs, float* hout1, long long ho1bs,
    float* cout0, long long co0bs, float* cout1, long long co1bs) {
  __shared__ float ws[360];   // [i|o|g|f][tap*10+ch], h-part weights (c = 256..265)
  const int b   = blockIdx.y / 10;
  const int hid = blockIdx.y % 10;
  for (int t = threadIdx.x; t < 360; t += 256) {
    int g = t / 90, t2 = t % 90;
    int tap = t2 / 10, ch = t2 % 10;
    float v;
    if (g < 3) v = gate_w[((size_t)(g * 10 + hid) * 266 + 256 + ch) * 9 + tap];
    else       v = f_w  [((size_t)hid * 266 + 256 + ch) * 9 + tap];
    ws[t] = v;
  }
  __syncthreads();
  const int s = blockIdx.x * 256 + threadIdx.x;   // 0..9215
  const int y = s / 96, x = s % 96;
  const size_t gb = ((size_t)(b * HH + y) * WW + x) * gstride;
  float ai = G[gb + gate_off + hid];
  float ao = G[gb + gate_off + 10 + hid];
  float ag = G[gb + gate_off + 20 + hid];
  const float* hs = hsum + (size_t)b * hsum_bs;
  for (int tap = 0; tap < 9; ++tap) {
    int iy = y + tap / 3 - 1, ix = x + tap % 3 - 1;
    if (iy < 0 || iy >= 96 || ix < 0 || ix >= 96) continue;
    int off = iy * 96 + ix;
    for (int ch = 0; ch < 10; ++ch) {
      float v = hs[ch * HW + off] * hsum_scale;
      ai += ws[tap * 10 + ch] * v;
      ao += ws[90 + tap * 10 + ch] * v;
      ag += ws[180 + tap * 10 + ch] * v;
    }
  }
  float ii = sigm(ai), oo = sigm(ao), gg = tanhf(ag);
  float c = ii * gg;
  const float* hc[4] = {h0, h1, h2, h3};
  long long    hb[4] = {h0bs, h1bs, h2bs, h3bs};
  const float* cc[4] = {c0, c1, c2, c3};
  long long    cb[4] = {c0bs, c1bs, c2bs, c3bs};
  const float fx = G[gb + f_off + hid];
  for (int j = 0; j < nchild; ++j) {
    const float* hj = hc[j] + (size_t)b * hb[j];
    float fa = fx;
    for (int tap = 0; tap < 9; ++tap) {
      int iy = y + tap / 3 - 1, ix = x + tap % 3 - 1;
      if (iy < 0 || iy >= 96 || ix < 0 || ix >= 96) continue;
      int off = iy * 96 + ix;
      for (int ch = 0; ch < 10; ++ch)
        fa += ws[270 + tap * 10 + ch] * hj[ch * HW + off];
    }
    float cj = cc[j][(size_t)b * cb[j] + (size_t)hid * HW + y * 96 + x];
    c += sigm(fa) * cj;
  }
  float hh = oo * tanhf(c);
  const size_t po = (size_t)hid * HW + y * 96 + x;
  if (hout0) hout0[(size_t)b * ho0bs + po] = hh;
  if (hout1) hout1[(size_t)b * ho1bs + po] = hh;
  if (cout0) cout0[(size_t)b * co0bs + po] = c;
  if (cout1) cout1[(size_t)b * co1bs + po] = c;
}

// ---------------- host launch ----------------
extern "C" void kernel_launch(void* const* d_in, const int* in_sizes, int n_in,
                              void* d_out, int out_size, void* d_ws, size_t ws_size,
                              hipStream_t stream) {
  (void)in_sizes; (void)n_in; (void)out_size; (void)ws_size;
  const float* p_fea = (const float*)d_in[3];
  const float* h_fea = (const float*)d_in[4];
  const float* f_fea = (const float*)d_in[5];
  const float* leaf_w      = (const float*)d_in[6];
  const float* gate_up_w   = (const float*)d_in[7];
  const float* f_up_w      = (const float*)d_in[8];
  const float* gate_lo_w   = (const float*)d_in[9];
  const float* f_lo_w      = (const float*)d_in[10];
  const float* gate_full_w = (const float*)d_in[11];
  const float* f_full_w    = (const float*)d_in[12];
  float* out = (float*)d_out;
  char* ws = (char*)d_ws;

  // workspace layout (bytes)
  __bf16* A_p = (__bf16*)(ws + 0);                       // 37,748,736
  __bf16* A_h = (__bf16*)(ws + 37748736);
  __bf16* A_f = (__bf16*)(ws + 75497472);
  __bf16* Wpl_p = (__bf16*)(ws + 113246208);             // 1,105,920
  __bf16* Wpl_h = (__bf16*)(ws + 114352128);             // 368,640
  __bf16* Wpl_f = (__bf16*)(ws + 114720768);             // 221,184
  __bf16* Wpk_p = (__bf16*)(ws + 114941952);
  __bf16* Wpk_h = (__bf16*)(ws + 116047872);
  __bf16* Wpk_f = (__bf16*)(ws + 116416512);
  float*  G_leaf = (float*)(ws + 116637696);             // 70,778,880
  float*  G_h    = (float*)(ws + 187416576);             // 23,592,960
  float*  G_f    = (float*)(ws + 211009536);             // 14,155,776
  float*  hsum_up = (float*)(ws + 225165312);            // 2,949,120
  float*  hsum_lo = (float*)(ws + 228114432);            // 2,949,120

  const int TPB = 256;
  // 1) activations -> NHWC bf16
  {
    int total = BB * SPB;
    int g = (total + TPB - 1) / TPB;
    nchw_to_nhwc_bf16<<<g, TPB, 0, stream>>>(p_fea, A_p, total);
    nchw_to_nhwc_bf16<<<g, TPB, 0, stream>>>(h_fea, A_h, total);
    nchw_to_nhwc_bf16<<<g, TPB, 0, stream>>>(f_fea, A_f, total);
  }
  // 2) pack + repack weights
  {
    int tp = 9 * CIN * 240, th = 9 * CIN * 80, tf = 9 * CIN * 48;
    pack_plain<<<(tp + TPB - 1) / TPB, TPB, 0, stream>>>(leaf_w, nullptr, nullptr, nullptr, Wpl_p, 240, 0, tp);
    pack_plain<<<(th + TPB - 1) / TPB, TPB, 0, stream>>>(gate_up_w, f_up_w, gate_lo_w, f_lo_w, Wpl_h, 80, 1, th);
    pack_plain<<<(tf + TPB - 1) / TPB, TPB, 0, stream>>>(gate_full_w, f_full_w, nullptr, nullptr, Wpl_f, 48, 2, tf);
    repack_bfrag<<<(tp + TPB - 1) / TPB, TPB, 0, stream>>>(Wpl_p, Wpk_p, 15, 240, tp);
    repack_bfrag<<<(th + TPB - 1) / TPB, TPB, 0, stream>>>(Wpl_h, Wpk_h, 5, 80, th);
    repack_bfrag<<<(tf + TPB - 1) / TPB, TPB, 0, stream>>>(Wpl_f, Wpk_f, 3, 48, tf);
  }
  // 3) leaf conv (WMMA): NT=15 = 3 passes x NB=5 -> 5 WMMAs per A-fragment pair
  conv_wmma<5><<<dim3(576, 3), TPB, 0, stream>>>(A_p, Wpk_p, G_leaf, 15, 240);
  {
    int total = BB * 6 * 10 * HW;
    leaf_epilogue<<<(total + TPB - 1) / TPB, TPB, 0, stream>>>(G_leaf, out, total);
  }
  // 4) child sums for up (children 0..3) and lo (children 4..5)
  {
    int total = BB * 10 * HW;
    sum_children<<<(total + TPB - 1) / TPB, TPB, 0, stream>>>(out + O_HALF_H, hsum_up, 4, total);
    sum_children<<<(total + TPB - 1) / TPB, TPB, 0, stream>>>(out + O_HALF_H + 4 * 92160, hsum_lo, 2, total);
  }
  // 5) shared x-part convs: h path NT=5 in ONE pass (NB=5); f path NT=3 in one pass (NB=3)
  conv_wmma<5><<<dim3(576, 1), TPB, 0, stream>>>(A_h, Wpk_h, G_h, 5, 80);
  conv_wmma<3><<<dim3(576, 1), TPB, 0, stream>>>(A_f, Wpk_f, G_f, 3, 48);

  const float* half_h = out + O_HALF_H;      // [8,6,10,96,96]
  const float* half_c = out + O_HALF_C;
  float* half2_h = out + O_HALF2_H;          // [8,2,10,96,96]
  float* half2_c = out + O_HALF2_C;

  // 6) upper tree cell: children 0..3; only c output used (-> half2_c slot 0)
  tree_epilogue<<<dim3(36, 80), TPB, 0, stream>>>(
      G_h, 80, 0, 30, gate_up_w, f_up_w,
      hsum_up, 92160LL, 1.0f,
      half_h + 0 * 92160, 552960LL, half_h + 1 * 92160, 552960LL,
      half_h + 2 * 92160, 552960LL, half_h + 3 * 92160, 552960LL,
      half_c + 0 * 92160, 552960LL, half_c + 1 * 92160, 552960LL,
      half_c + 2 * 92160, 552960LL, half_c + 3 * 92160, 552960LL,
      4,
      nullptr, 0LL, nullptr, 0LL,
      half2_c + 0 * 92160, 184320LL, nullptr, 0LL);
  // 7) lower tree cell: children 4,5; h -> half2_h slots 0 & 1, c -> half2_c slot 1
  tree_epilogue<<<dim3(36, 80), TPB, 0, stream>>>(
      G_h, 80, 40, 70, gate_lo_w, f_lo_w,
      hsum_lo, 92160LL, 1.0f,
      half_h + 4 * 92160, 552960LL, half_h + 5 * 92160, 552960LL,
      nullptr, 0LL, nullptr, 0LL,
      half_c + 4 * 92160, 552960LL, half_c + 5 * 92160, 552960LL,
      nullptr, 0LL, nullptr, 0LL,
      2,
      half2_h + 0 * 92160, 184320LL, half2_h + 1 * 92160, 184320LL,
      half2_c + 1 * 92160, 184320LL, nullptr, 0LL);
  // 8) full tree cell: h children both = lower h (hsum = 2*lower h); c children = [upper c, lower c]
  tree_epilogue<<<dim3(36, 80), TPB, 0, stream>>>(
      G_f, 48, 0, 30, gate_full_w, f_full_w,
      half2_h + 0 * 92160, 184320LL, 2.0f,
      half2_h + 0 * 92160, 184320LL, half2_h + 0 * 92160, 184320LL,
      nullptr, 0LL, nullptr, 0LL,
      half2_c + 0 * 92160, 184320LL, half2_c + 1 * 92160, 184320LL,
      nullptr, 0LL, nullptr, 0LL,
      2,
      out + O_FULL_H, 92160LL, nullptr, 0LL,
      out + O_FULL_C, 92160LL, nullptr, 0LL);
}
